// VulnGAT_31241592111613
// MI455X (gfx1250) — matrix operand order repaired and checked by reading, not compiled, not run on hardware
//
#include <hip/hip_runtime.h>
#include <hip/hip_fp16.h>

#define NN   20000
#define EE   320000
#define GG   64
#define FINd 22
#define HIDd 128
#define ETOT (EE + NN)

typedef __attribute__((ext_vector_type(16))) _Float16 v16h;
typedef __attribute__((ext_vector_type(8)))  float    v8f;

// ---------- helpers ----------
__device__ __forceinline__ unsigned int encf(float f) {
  unsigned int u = __float_as_uint(f);
  return (u & 0x80000000u) ? ~u : (u | 0x80000000u);
}
__device__ __forceinline__ float decf(unsigned int u) {
  u = (u & 0x80000000u) ? (u & 0x7FFFFFFFu) : ~u;
  return __uint_as_float(u);
}
union H16 { _Float16 h; unsigned short u; };
__device__ __forceinline__ unsigned short f2h_bits(float f) {
  H16 x; x.h = (_Float16)f; return x.u;
}
#define ENC_NEG_INF 0x007FFFFFu   // encf(-inf)

// ---------- generic fill ----------
__global__ void fill_u32(unsigned int* p, unsigned int v, int n) {
  int i = blockIdx.x * blockDim.x + threadIdx.x;
  if (i < n) p[i] = v;
}

// ---------- edge-attr mean (counts of 3 categories) ----------
__global__ void attr_count_kernel(const int* __restrict__ eattr, unsigned int* cnt) {
  int e = blockIdx.x * blockDim.x + threadIdx.x;
  if (e < EE) atomicAdd(&cnt[eattr[e] & 3], 1u);
}
__global__ void emean_fin_kernel(const unsigned int* cnt, const float* __restrict__ emb,
                                 float* emean) {
  int d = threadIdx.x;
  if (d >= 8) return;
  float s = (float)cnt[0] * emb[d] + (float)cnt[1] * emb[8 + d] + (float)cnt[2] * emb[16 + d];
  emean[d] = s / (float)EE;
}

// ---------- input layer: h = relu(bn(x@w_in + b_in)); also f16 copy ----------
__global__ void input_kernel(const float* __restrict__ x, const float* __restrict__ w_in,
                             const float* __restrict__ b_in, const float* __restrict__ g_in,
                             const float* __restrict__ be_in,
                             float* __restrict__ h, unsigned short* __restrict__ hf16) {
  int n = blockIdx.x, c = threadIdx.x;   // 128 threads
  __shared__ float xs[FINd];
  if (c < FINd) xs[c] = x[n * FINd + c];
  __syncthreads();
  float acc = b_in[c];
#pragma unroll
  for (int k = 0; k < FINd; ++k) acc = fmaf(xs[k], w_in[k * HIDd + c], acc);
  const float inv = rsqrtf(1.0f + 1e-5f);
  float v = acc * (g_in[c] * inv) + be_in[c];
  v = fmaxf(v, 0.0f);
  h[n * HIDd + c]    = v;
  hf16[n * HIDd + c] = f2h_bits(v);
}

// ---------- pack W [128 x dout] f32 row-major -> WMMA-B f16 fragments ----------
// layout: idx = ((kk*nct + nt)*32 + lane)*8 + j  holds {K=kb+2(j&3), K=kb+2(j&3)+1} of col
//         col = nt*16 + (lane&15), kb = kk*32 + 8*(lane>>4) + 16*(j>>2)
__global__ void pack_w_kernel(const float* __restrict__ W, unsigned int* __restrict__ Bpk,
                              int dout) {
  int idx = blockIdx.x * blockDim.x + threadIdx.x;
  int nct = dout >> 4;
  int total = 1024 * nct;                // 4 * nct * 32 * 8
  if (idx >= total) return;
  int j    = idx & 7;
  int lane = (idx >> 3) & 31;
  int t    = idx >> 8;
  int nt   = t % nct;
  int kk   = t / nct;
  int col  = nt * 16 + (lane & 15);
  int k    = kk * 32 + ((lane >> 4) << 3) + ((j >> 2) << 4) + 2 * (j & 3);
  unsigned int lo = f2h_bits(W[k * dout + col]);
  unsigned int hi = f2h_bits(W[(k + 1) * dout + col]);
  Bpk[idx] = lo | (hi << 16);
}

// ---------- WMMA GEMM: out[N,dout] = f16(A[N,128]) @ f16(Bpk) + bias ----------
// grid.x = N/16 row tiles, blockDim = 32*nct (one wave per 16-col tile)
__global__ void gemm_wmma_kernel(const unsigned short* __restrict__ Af16,
                                 const unsigned int* __restrict__ Bpk,
                                 const float* __restrict__ bias,
                                 float* __restrict__ out, int dout) {
  const int nct  = dout >> 4;
  const int nt   = threadIdx.x >> 5;     // wave id == column tile
  const int lane = threadIdx.x & 31;
  const int rowbase = blockIdx.x * 16;

  const int arow  = rowbase + (lane & 15);
  const int khalf = (lane >> 4) * 8;     // A layout: lanes 0-15 K0-7/16-23; 16-31 K8-15/24-31

  v8f acc = {};
  union { uint4 u[2]; v16h v; } a, b;
#pragma unroll
  for (int kk = 0; kk < 4; ++kk) {
    const unsigned short* ap = Af16 + arow * HIDd + kk * 32 + khalf;
    a.u[0] = *reinterpret_cast<const uint4*>(ap);
    a.u[1] = *reinterpret_cast<const uint4*>(ap + 16);
    const unsigned int* bp = Bpk + (((kk * nct + nt) * 32) + lane) * 8;
    b.u[0] = *reinterpret_cast<const uint4*>(bp);
    b.u[1] = *reinterpret_cast<const uint4*>(bp + 4);
    acc = __builtin_amdgcn_wmma_f32_16x16x32_f16(false, a.v, false, b.v,
                                                 (short)0, acc, false, false);
  }
  const int r0  = rowbase + (lane >> 4) * 8;   // D layout: VGPR j -> row j (+8 for hi lanes)
  const int col = nt * 16 + (lane & 15);
  const float bc = bias[col];
#pragma unroll
  for (int j = 0; j < 8; ++j)
    out[(r0 + j) * dout + col] = acc[j] + bc;
}

// ---------- edge pass 1: logits + segment max (wave per edge) ----------
template <int DOUT, int H>
__global__ void edge_logits_kernel(const int* __restrict__ eidx, const int* __restrict__ eattr,
                                   const float* __restrict__ emb, const float* __restrict__ emean,
                                   const float* __restrict__ We, const float* __restrict__ att,
                                   const float* __restrict__ xl, const float* __restrict__ xr,
                                   float* __restrict__ lg, unsigned int* __restrict__ mxe) {
  const int lane = threadIdx.x & 31;
  const int e = blockIdx.x * (blockDim.x >> 5) + (threadIdx.x >> 5);
  if (e >= ETOT) return;
  int src, dst;
  float e8[8];
  if (e < EE) {
    src = eidx[e]; dst = eidx[EE + e];
    int a = eattr[e] & 3;
#pragma unroll
    for (int d = 0; d < 8; ++d) e8[d] = emb[a * 8 + d];
  } else {
    src = dst = e - EE;
#pragma unroll
    for (int d = 0; d < 8; ++d) e8[d] = emean[d];
  }
  constexpr int CPL = DOUT / 32;         // channels per lane
  float partial = 0.0f;
#pragma unroll
  for (int cc = 0; cc < CPL; ++cc) {
    int c = lane * CPL + cc;
    float ee = 0.0f;
#pragma unroll
    for (int k = 0; k < 8; ++k) ee = fmaf(e8[k], We[k * DOUT + c], ee);
    float m = xl[src * DOUT + c] + xr[dst * DOUT + c] + ee;
    m = (m > 0.0f) ? m : 0.2f * m;       // leaky_relu(0.2)
    partial = fmaf(att[c], m, partial);
  }
  constexpr int LPH = 32 / H;            // lanes per head
#pragma unroll
  for (int s = 1; s < LPH; s <<= 1) partial += __shfl_xor(partial, s, 32);
  if ((lane % LPH) == 0) {
    int hh = lane / LPH;
    lg[e * H + hh] = partial;
    atomicMax(&mxe[dst * H + hh], encf(partial));
  }
}

// ---------- edge pass 2: a = exp(lg - max); segment sum ----------
__global__ void edge_alpha_kernel(const int* __restrict__ eidx, const float* __restrict__ lg,
                                  const unsigned int* __restrict__ mxe,
                                  float* __restrict__ aexp, float* __restrict__ den, int H) {
  int i = blockIdx.x * blockDim.x + threadIdx.x;
  if (i >= ETOT * H) return;
  int e = i / H, hh = i - e * H;
  int dst = (e < EE) ? eidx[EE + e] : (e - EE);
  float a = __expf(lg[i] - decf(mxe[dst * H + hh]));
  aexp[i] = a;
  atomicAdd(&den[dst * H + hh], a);
}

// ---------- edge pass 3: agg[dst] += xl[src] * alpha (wave per edge) ----------
template <int DOUT, int H>
__global__ void edge_aggr_kernel(const int* __restrict__ eidx, const float* __restrict__ xl,
                                 const float* __restrict__ aexp, const float* __restrict__ den,
                                 float* __restrict__ agg) {
  const int lane = threadIdx.x & 31;
  const int e = blockIdx.x * (blockDim.x >> 5) + (threadIdx.x >> 5);
  if (e >= ETOT) return;
  int src, dst;
  if (e < EE) { src = eidx[e]; dst = eidx[EE + e]; }
  else        { src = dst = e - EE; }
  constexpr int CPL = DOUT / 32;
#pragma unroll
  for (int cc = 0; cc < CPL; ++cc) {
    int c = lane * CPL + cc;
    int hh = c >> 5;
    float alpha = aexp[e * H + hh] / (den[dst * H + hh] + 1e-16f);
    atomicAdd(&agg[dst * DOUT + c], xl[src * DOUT + c] * alpha);
  }
}

// ---------- post: h = (residual? h : 0) + elu(bn(agg + bo)); refresh f16 ----------
__global__ void post_kernel(const float* __restrict__ agg, const float* __restrict__ bo,
                            const float* __restrict__ g, const float* __restrict__ bb,
                            float* __restrict__ hio, unsigned short* __restrict__ hf16,
                            int dout, int total, int residual) {
  int i = blockIdx.x * blockDim.x + threadIdx.x;
  if (i >= total) return;
  int c = i % dout;
  const float inv = rsqrtf(1.0f + 1e-5f);
  float v = agg[i] + bo[c];
  v = v * (g[c] * inv) + bb[c];
  v = (v > 0.0f) ? v : (__expf(v) - 1.0f);   // elu
  if (residual) v += hio[i];
  hio[i] = v;
  if (hf16) hf16[i] = f2h_bits(v);
}

// ---------- pooling over batch ----------
__global__ void pool_kernel(const float* __restrict__ h3, const int* __restrict__ batch,
                            float* __restrict__ psum, unsigned int* __restrict__ pmax,
                            float* __restrict__ pcnt) {
  int i = blockIdx.x * blockDim.x + threadIdx.x;
  if (i >= NN * 64) return;
  int n = i >> 6, c = i & 63;
  int b = batch[n];
  float v = h3[i];
  atomicAdd(&psum[b * 64 + c], v);
  atomicMax(&pmax[b * 64 + c], encf(v));
  if (c == 0) atomicAdd(&pcnt[b], 1.0f);
}

// ---------- classifier head: [G,128] -> relu(@wc1+bc1) -> @wc2+bc2 ----------
__global__ void head_kernel(const float* __restrict__ psum, const unsigned int* __restrict__ pmax,
                            const float* __restrict__ pcnt,
                            const float* __restrict__ wc1, const float* __restrict__ bc1,
                            const float* __restrict__ wc2, const float* __restrict__ bc2,
                            float* __restrict__ out) {
  int g = blockIdx.x, t = threadIdx.x;   // 64 threads
  __shared__ float p[128];
  __shared__ float hid[64];
  float cnt = fmaxf(pcnt[g], 1.0f);
  p[t] = psum[g * 64 + t] / cnt;
  float m = decf(pmax[g * 64 + t]);
  p[64 + t] = (m < -3.0e38f) ? 0.0f : m;
  __syncthreads();
  float acc = bc1[t];
#pragma unroll 8
  for (int k = 0; k < 128; ++k) acc = fmaf(p[k], wc1[k * 64 + t], acc);
  hid[t] = fmaxf(acc, 0.0f);
  __syncthreads();
  if (t < 5) {
    float o = bc2[t];
#pragma unroll 8
    for (int j = 0; j < 64; ++j) o = fmaf(hid[j], wc2[j * 5 + t], o);
    out[g * 5 + t] = o;
  }
}

// =====================================================================
extern "C" void kernel_launch(void* const* d_in, const int* in_sizes, int n_in,
                              void* d_out, int out_size, void* d_ws, size_t ws_size,
                              hipStream_t stream) {
  (void)in_sizes; (void)n_in; (void)out_size; (void)ws_size;

  const float* x     = (const float*)d_in[0];
  const int*   eidx  = (const int*)d_in[1];
  const int*   eattr = (const int*)d_in[2];
  const int*   batch = (const int*)d_in[3];
  const float* emb   = (const float*)d_in[4];
  const float* w_in  = (const float*)d_in[5];
  const float* b_in  = (const float*)d_in[6];
  const float* g_in  = (const float*)d_in[7];
  const float* be_in = (const float*)d_in[8];
  const float *wl[3], *bl[3], *wr[3], *br[3], *we[3], *att[3], *bo[3], *gl[3], *bb[3];
  for (int l = 0; l < 3; ++l) {
    int base = 9 + 9 * l;
    wl[l]  = (const float*)d_in[base + 0];
    bl[l]  = (const float*)d_in[base + 1];
    wr[l]  = (const float*)d_in[base + 2];
    br[l]  = (const float*)d_in[base + 3];
    we[l]  = (const float*)d_in[base + 4];
    att[l] = (const float*)d_in[base + 5];
    bo[l]  = (const float*)d_in[base + 6];
    gl[l]  = (const float*)d_in[base + 7];
    bb[l]  = (const float*)d_in[base + 8];
  }
  const float* wc1 = (const float*)d_in[36];
  const float* bc1 = (const float*)d_in[37];
  const float* wc2 = (const float*)d_in[38];
  const float* bc2 = (const float*)d_in[39];

  // ---- workspace layout (bump allocator, 256B aligned) ----
  char* ws = (char*)d_ws;
  size_t off = 0;
  auto alloc = [&](size_t bytes) -> void* {
    void* p = ws + off;
    off += (bytes + 255) & ~(size_t)255;
    return p;
  };
  float*          h    = (float*)alloc(sizeof(float) * NN * HIDd);
  unsigned short* hf16 = (unsigned short*)alloc(sizeof(unsigned short) * NN * HIDd);
  float*          xl   = (float*)alloc(sizeof(float) * NN * HIDd);
  float*          xr   = (float*)alloc(sizeof(float) * NN * HIDd);
  float*          agg  = (float*)alloc(sizeof(float) * NN * HIDd);
  float*          lg   = (float*)alloc(sizeof(float) * (size_t)ETOT * 4);
  float*          aexp = (float*)alloc(sizeof(float) * (size_t)ETOT * 4);
  unsigned int*   mxe  = (unsigned int*)alloc(sizeof(unsigned int) * NN * 4);
  float*          den  = (float*)alloc(sizeof(float) * NN * 4);
  unsigned int*   acnt = (unsigned int*)alloc(sizeof(unsigned int) * 4);
  float*          emean = (float*)alloc(sizeof(float) * 8);
  const int douts[3] = {128, 128, 64};
  unsigned int* wpack[6];
  for (int l = 0; l < 3; ++l) {
    wpack[2 * l]     = (unsigned int*)alloc(sizeof(unsigned int) * 64 * douts[l]);
    wpack[2 * l + 1] = (unsigned int*)alloc(sizeof(unsigned int) * 64 * douts[l]);
  }
  float*        psum = (float*)alloc(sizeof(float) * GG * 64);
  unsigned int* pmax = (unsigned int*)alloc(sizeof(unsigned int) * GG * 64);
  float*        pcnt = (float*)alloc(sizeof(float) * GG);

  auto cdiv = [](int a, int b) { return (a + b - 1) / b; };

  // ---- pack weights into WMMA-B fragment layout ----
  for (int l = 0; l < 3; ++l) {
    int tot = 1024 * (douts[l] >> 4);
    pack_w_kernel<<<cdiv(tot, 256), 256, 0, stream>>>(wl[l], wpack[2 * l], douts[l]);
    pack_w_kernel<<<cdiv(tot, 256), 256, 0, stream>>>(wr[l], wpack[2 * l + 1], douts[l]);
  }

  // ---- edge-attr mean for self loops ----
  fill_u32<<<1, 32, 0, stream>>>(acnt, 0u, 4);
  attr_count_kernel<<<cdiv(EE, 256), 256, 0, stream>>>(eattr, acnt);
  emean_fin_kernel<<<1, 8, 0, stream>>>(acnt, emb, emean);

  // ---- input layer ----
  input_kernel<<<NN, 128, 0, stream>>>(x, w_in, b_in, g_in, be_in, h, hf16);

  // ---- GAT layers ----
  const int eb = cdiv(ETOT, 8);   // 8 waves (edges) per 256-thread block
  for (int l = 0; l < 3; ++l) {
    const int dout = douts[l];
    const int H = (l < 2) ? 4 : 2;
    const int nct = dout >> 4;

    gemm_wmma_kernel<<<NN / 16, 32 * nct, 0, stream>>>(hf16, wpack[2 * l],     bl[l], xl, dout);
    gemm_wmma_kernel<<<NN / 16, 32 * nct, 0, stream>>>(hf16, wpack[2 * l + 1], br[l], xr, dout);

    fill_u32<<<cdiv(NN * H, 256), 256, 0, stream>>>(mxe, ENC_NEG_INF, NN * H);
    fill_u32<<<cdiv(NN * H, 256), 256, 0, stream>>>((unsigned int*)den, 0u, NN * H);
    fill_u32<<<cdiv(NN * dout, 256), 256, 0, stream>>>((unsigned int*)agg, 0u, NN * dout);

    if (l < 2) {
      edge_logits_kernel<128, 4><<<eb, 256, 0, stream>>>(eidx, eattr, emb, emean, we[l],
                                                         att[l], xl, xr, lg, mxe);
    } else {
      edge_logits_kernel<64, 2><<<eb, 256, 0, stream>>>(eidx, eattr, emb, emean, we[l],
                                                        att[l], xl, xr, lg, mxe);
    }
    edge_alpha_kernel<<<cdiv(ETOT * H, 256), 256, 0, stream>>>(eidx, lg, mxe, aexp, den, H);
    if (l < 2) {
      edge_aggr_kernel<128, 4><<<eb, 256, 0, stream>>>(eidx, xl, aexp, den, agg);
      post_kernel<<<cdiv(NN * dout, 256), 256, 0, stream>>>(agg, bo[l], gl[l], bb[l],
                                                            h, hf16, dout, NN * dout, 1);
    } else {
      edge_aggr_kernel<64, 2><<<eb, 256, 0, stream>>>(eidx, xl, aexp, den, agg);
      // no residual on layer 3; write h3 into xl buffer (no f16 copy needed)
      post_kernel<<<cdiv(NN * dout, 256), 256, 0, stream>>>(agg, bo[l], gl[l], bb[l],
                                                            xl, (unsigned short*)nullptr,
                                                            dout, NN * dout, 0);
    }
  }

  // ---- pooling + classifier head ----
  fill_u32<<<cdiv(GG * 64, 256), 256, 0, stream>>>((unsigned int*)psum, 0u, GG * 64);
  fill_u32<<<cdiv(GG * 64, 256), 256, 0, stream>>>(pmax, ENC_NEG_INF, GG * 64);
  fill_u32<<<1, 64, 0, stream>>>((unsigned int*)pcnt, 0u, GG);
  pool_kernel<<<cdiv(NN * 64, 256), 256, 0, stream>>>(xl, batch, psum, pmax, pcnt);
  head_kernel<<<GG, 64, 0, stream>>>(psum, pmax, pcnt, wc1, bc1, wc2, bc2, (float*)d_out);
}
